// MultiHeadAttention_47236050321945
// MI455X (gfx1250) — matrix operand order repaired and checked by reading, not compiled, not run on hardware
//
#include <hip/hip_runtime.h>
#include <hip/hip_bf16.h>
#include <stdint.h>

// ---------------- problem constants (from reference) ----------------
#define BB   4
#define SS   2048
#define DD   1024
#define HH   16
#define HDD  64
#define MM   (BB*SS)          // 8192 rows of X

typedef __attribute__((ext_vector_type(16))) __bf16 v16bf;
typedef __attribute__((ext_vector_type(8)))  float  v8f;
typedef __attribute__((ext_vector_type(4)))  unsigned int v4u;
typedef __attribute__((ext_vector_type(8)))  int v8i;
typedef __attribute__((ext_vector_type(4)))  int v4i;

union FragU { v16bf v; uint4 q[2]; };

__device__ __forceinline__ unsigned short f2bf(float f) {
  union { float f; unsigned u; } x; x.f = f;
  unsigned u = x.u;
  u += 0x7FFFu + ((u >> 16) & 1u);   // round to nearest even
  return (unsigned short)(u >> 16);
}

// A fragment (16x32 bf16), CDNA5 layout:
// lanes 0-15: m=lane, V0-3 = K[k0..k0+7], V4-7 = K[k0+16..k0+23]
// lanes 16-31: m=lane-16, V0-3 = K[k0+8..15], V4-7 = K[k0+24..31]
__device__ __forceinline__ v16bf load_frag_a(const unsigned short* base, int lda,
                                             int row0, int k0, int lane) {
  int m = lane & 15, hi = lane >> 4;
  const unsigned short* p = base + (size_t)(row0 + m) * lda + k0;
  FragU f;
  f.q[0] = *(const uint4*)(p + hi * 8);
  f.q[1] = *(const uint4*)(p + 16 + hi * 8);
  return f.v;
}

// B fragment (32x16 bf16) where B(k,n) = W[n0+n][k0+k] (W row-major, stride ldw).
// lane = column n; lanes 0-15 hold K=0..7,16..23, lanes 16-31 hold K=8..15,24..31 (matches A).
__device__ __forceinline__ v16bf load_frag_b(const unsigned short* base, int ldw,
                                             int n0, int k0, int lane) {
  int n = lane & 15, hi = lane >> 4;
  const unsigned short* p = base + (size_t)(n0 + n) * ldw + k0;
  FragU f;
  f.q[0] = *(const uint4*)(p + hi * 8);
  f.q[1] = *(const uint4*)(p + 16 + hi * 8);
  return f.v;
}

// Issue one TDM DMA: stage `rows` x DD bf16 tile (row-major, stride DD) from
// global `gptr` into LDS at `ldsPtr`, then wait for completion.
// D# per CDNA5 ISA (2-D tensor, groups 2/3 zero).
__device__ __forceinline__ void tdm_stage_rows(const unsigned short* gptr,
                                               unsigned short* ldsPtr, int rows) {
  unsigned long long ga = (unsigned long long)(size_t)gptr;
  unsigned lds_base = (unsigned)(size_t)ldsPtr;
  v4u g0;
  g0[0] = 1u;                                        // count=1, user mode
  g0[1] = lds_base;                                  // lds_addr
  g0[2] = (unsigned)(ga & 0xFFFFFFFFull);            // global_addr[31:0]
  g0[3] = (unsigned)((ga >> 32) & 0x1FFFFFFull)      // global_addr[56:32]
        | (2u << 30);                                // type = 2 ("image")
  v8i g1;
  g1[0] = (int)(1u << 16);                           // data_size = 1 (2 bytes)
  g1[1] = (int)((unsigned)DD << 16);                 // tensor_dim0 = 1024
  g1[2] = (int)((unsigned)MM << 16);                 // tensor_dim1 = 8192
  g1[3] = (int)((unsigned)DD << 16);                 // tile_dim0 = 1024
  g1[4] = rows;                                      // tile_dim1
  g1[5] = DD;                                        // tensor_dim0_stride
  g1[6] = 0;
  g1[7] = 0;
  v4i gz = {0, 0, 0, 0};
#if __clang_major__ >= 23
  v8i gz8 = {0, 0, 0, 0, 0, 0, 0, 0};
  __builtin_amdgcn_tensor_load_to_lds(g0, g1, gz, gz, gz8, 0);
#else
  __builtin_amdgcn_tensor_load_to_lds(g0, g1, gz, gz, 0);
#endif
  __builtin_amdgcn_s_wait_tensorcnt(0);
}

// ---------------- fp32 -> bf16 conversion ----------------
__global__ void cvt_bf16_kernel(const float* __restrict__ in,
                                unsigned short* __restrict__ out, int n) {
  int i = blockIdx.x * blockDim.x + threadIdx.x;
  if (i < n) out[i] = f2bf(in[i]);
}

// ---------------- generic wave-tiled bf16 GEMM: C = A * W^T ----------------
// A: MM x DD bf16 row-major.  W: DD x DD bf16 row-major (out,in).
// All 8 waves of a block share one 32-row A strip -> TDM-stage it into LDS.
// mode 0: out bf16, head-split (B,H,S,HD)             (Q, K)
// mode 1: out bf16, head-split transposed (B,H,HD,S)  (V^T)
// mode 2: out fp32 row-major MM x DD                  (final projection)
__global__ void __launch_bounds__(256)
gemm_bf16_kernel(const unsigned short* __restrict__ A,
                 const unsigned short* __restrict__ W,
                 void* __restrict__ out, int mode) {
  __shared__ unsigned short Atile[32 * DD];   // 64 KB A strip (rows m0..m0+31)
  const int K = DD, N = DD;
  int lane = threadIdx.x & 31;
  int wave = threadIdx.x >> 5;
  int tile = blockIdx.x * 8 + wave;          // 8 waves / block
  const int tilesN = N / 32;                 // 32 -> all 8 waves share m0
  int m0 = (tile / tilesN) * 32;
  int n0 = (tile % tilesN) * 32;

  if (wave == 0)
    tdm_stage_rows(A + (size_t)m0 * K, Atile, 32);
  __syncthreads();

  v8f acc00 = {}, acc01 = {}, acc10 = {}, acc11 = {};
  for (int k0 = 0; k0 < K; k0 += 32) {
    v16bf a0 = load_frag_a(Atile, K, 0,  k0, lane);   // LDS (ds_load_b128)
    v16bf a1 = load_frag_a(Atile, K, 16, k0, lane);
    v16bf b0 = load_frag_b(W, K, n0,      k0, lane);  // global stream
    v16bf b1 = load_frag_b(W, K, n0 + 16, k0, lane);
    acc00 = __builtin_amdgcn_wmma_f32_16x16x32_bf16(false, a0, false, b0, (short)0, acc00, false, false);
    acc01 = __builtin_amdgcn_wmma_f32_16x16x32_bf16(false, a0, false, b1, (short)0, acc01, false, false);
    acc10 = __builtin_amdgcn_wmma_f32_16x16x32_bf16(false, a1, false, b0, (short)0, acc10, false, false);
    acc11 = __builtin_amdgcn_wmma_f32_16x16x32_bf16(false, a1, false, b1, (short)0, acc11, false, false);
  }

  int hi = lane >> 4, nl = lane & 15;
#pragma unroll
  for (int mi = 0; mi < 2; ++mi) {
#pragma unroll
    for (int ni = 0; ni < 2; ++ni) {
      v8f acc = (mi == 0) ? (ni == 0 ? acc00 : acc01)
                          : (ni == 0 ? acc10 : acc11);
#pragma unroll
      for (int v = 0; v < 8; ++v) {
        int m = m0 + mi * 16 + v + 8 * hi;   // C layout: row = vgpr + 8*half
        int n = n0 + ni * 16 + nl;           // col = lane % 16
        float val = acc[v];
        if (mode == 2) {
          ((float*)out)[(size_t)m * N + n] = val;
        } else {
          int b = m >> 11, s = m & (SS - 1);
          int h = n >> 6,  hd = n & (HDD - 1);
          size_t idx = (mode == 0)
            ? (((size_t)(b * HH + h) * SS) + s) * HDD + hd
            : (((size_t)(b * HH + h) * HDD) + hd) * SS + s;
          ((unsigned short*)out)[idx] = f2bf(val);
        }
      }
    }
  }
}

// ---------------- flash attention (causal, online softmax) ----------------
// Q,K: (B,H,S,HD) bf16.  Vt: (B,H,HD,S) bf16.  ctx out: (B,S,D) bf16.
// 4 waves / block; each wave owns 16 query rows, loops over 32-key chunks.
__global__ void __launch_bounds__(128)
attn_kernel(const unsigned short* __restrict__ Q,
            const unsigned short* __restrict__ Km,
            const unsigned short* __restrict__ Vt,
            unsigned short* __restrict__ ctx) {
  __shared__ unsigned short Pbuf[4][16 * 32];

  int lane = threadIdx.x & 31;
  int wave = threadIdx.x >> 5;
  int hi = lane >> 4, nl = lane & 15;
  int bh = blockIdx.y;
  int b = bh >> 4, h = bh & 15;
  int q0 = blockIdx.x * 64 + wave * 16;

  const unsigned short* Qh = Q  + (size_t)bh * SS * HDD;
  const unsigned short* Kh = Km + (size_t)bh * SS * HDD;
  const unsigned short* Vh = Vt + (size_t)bh * HDD * SS;
  unsigned short* Pb = Pbuf[wave];

  // Q rows for this wave, preloaded once (A fragments over d=0..31, 32..63)
  v16bf qa0 = load_frag_a(Qh, HDD, q0, 0,  lane);
  v16bf qa1 = load_frag_a(Qh, HDD, q0, 32, lane);

  v8f acc0 = {}, acc1 = {}, acc2 = {}, acc3 = {};
  float rmax[8], rsum[8];
#pragma unroll
  for (int v = 0; v < 8; ++v) { rmax[v] = -3.0e38f; rsum[v] = 0.f; }

  int nchunk = ((q0 + 15) >> 5) + 1;   // causal: keys 0 .. q0+15
  for (int c = 0; c < nchunk; ++c) {
    int j0 = c * 32;

    // scores: S = Q(16x64) * K_chunk^T, as two 16x16 tiles
    v8f s0 = {}, s1 = {};
    {
      v16bf kb0 = load_frag_b(Kh, HDD, j0, 0,  lane);
      v16bf kb1 = load_frag_b(Kh, HDD, j0, 32, lane);
      s0 = __builtin_amdgcn_wmma_f32_16x16x32_bf16(false, qa0, false, kb0, (short)0, s0, false, false);
      s0 = __builtin_amdgcn_wmma_f32_16x16x32_bf16(false, qa1, false, kb1, (short)0, s0, false, false);
      v16bf kb2 = load_frag_b(Kh, HDD, j0 + 16, 0,  lane);
      v16bf kb3 = load_frag_b(Kh, HDD, j0 + 16, 32, lane);
      s1 = __builtin_amdgcn_wmma_f32_16x16x32_bf16(false, qa0, false, kb2, (short)0, s1, false, false);
      s1 = __builtin_amdgcn_wmma_f32_16x16x32_bf16(false, qa1, false, kb3, (short)0, s1, false, false);
    }

    // online softmax in C layout (row m = v + 8*hi, col = nl; row spans 16 lanes)
#pragma unroll
    for (int v = 0; v < 8; ++v) {
      int mg = q0 + v + 8 * hi;
      float e0 = s0[v] * 0.125f;                  // 1/sqrt(64)
      float e1 = s1[v] * 0.125f;
      if (j0 + nl > mg)       e0 = -3.0e38f;      // causal mask
      if (j0 + 16 + nl > mg)  e1 = -3.0e38f;

      float t = fmaxf(e0, e1);
#pragma unroll
      for (int off = 1; off < 16; off <<= 1)
        t = fmaxf(t, __shfl_xor(t, off, 32));
      float nm = fmaxf(rmax[v], t);
      float alpha = __expf(rmax[v] - nm);
      rmax[v] = nm;
      float p0 = __expf(e0 - nm);
      float p1 = __expf(e1 - nm);
      float ps = p0 + p1;
#pragma unroll
      for (int off = 1; off < 16; off <<= 1)
        ps += __shfl_xor(ps, off, 32);
      rsum[v] = rsum[v] * alpha + ps;
      acc0[v] *= alpha; acc1[v] *= alpha; acc2[v] *= alpha; acc3[v] *= alpha;

      // stash P tile to LDS (row-major 16x32 bf16) to transpose C->A layout
      int mrow = v + 8 * hi;
      Pb[mrow * 32 + nl]      = f2bf(p0);
      Pb[mrow * 32 + 16 + nl] = f2bf(p1);
    }

    // wave-private LDS RAW: force all ds_stores complete before ds_loads
    asm volatile("s_wait_dscnt 0" ::: "memory");

    // P as a 16x32 A fragment straight from LDS
    v16bf pa;
    {
      const unsigned short* p = Pb + nl * 32;
      FragU f;
      f.q[0] = *(const uint4*)(p + hi * 8);
      f.q[1] = *(const uint4*)(p + 16 + hi * 8);
      pa = f.v;
    }

    // ctx += P(16x32) * V_chunk(32x64); B(k,n) = Vt[dtile*16+n][j0+k]
    v16bf vb0 = load_frag_b(Vh, SS, 0,  j0, lane);
    acc0 = __builtin_amdgcn_wmma_f32_16x16x32_bf16(false, pa, false, vb0, (short)0, acc0, false, false);
    v16bf vb1 = load_frag_b(Vh, SS, 16, j0, lane);
    acc1 = __builtin_amdgcn_wmma_f32_16x16x32_bf16(false, pa, false, vb1, (short)0, acc1, false, false);
    v16bf vb2 = load_frag_b(Vh, SS, 32, j0, lane);
    acc2 = __builtin_amdgcn_wmma_f32_16x16x32_bf16(false, pa, false, vb2, (short)0, acc2, false, false);
    v16bf vb3 = load_frag_b(Vh, SS, 48, j0, lane);
    acc3 = __builtin_amdgcn_wmma_f32_16x16x32_bf16(false, pa, false, vb3, (short)0, acc3, false, false);
  }

  // normalize and write ctx in (B,S,D) layout (heads re-interleaved)
#pragma unroll
  for (int v = 0; v < 8; ++v) {
    float inv = 1.0f / rsum[v];
    int srow = q0 + v + 8 * hi;
    size_t rowbase = ((size_t)b * SS + srow) * DD + (size_t)h * HDD;
    ctx[rowbase + 0  + nl] = f2bf(acc0[v] * inv);
    ctx[rowbase + 16 + nl] = f2bf(acc1[v] * inv);
    ctx[rowbase + 32 + nl] = f2bf(acc2[v] * inv);
    ctx[rowbase + 48 + nl] = f2bf(acc3[v] * inv);
  }
}

// ---------------- launch ----------------
extern "C" void kernel_launch(void* const* d_in, const int* in_sizes, int n_in,
                              void* d_out, int out_size, void* d_ws, size_t ws_size,
                              hipStream_t stream) {
  const float* x  = (const float*)d_in[0];
  const float* Wq = (const float*)d_in[1];
  const float* Wk = (const float*)d_in[2];
  const float* Wv = (const float*)d_in[3];
  const float* Wo = (const float*)d_in[4];

  const size_t XE = (size_t)MM * DD;   // 8,388,608
  const size_t WE = (size_t)DD * DD;   // 1,048,576

  unsigned short* ws  = (unsigned short*)d_ws;
  unsigned short* xbf = ws;
  unsigned short* wqb = xbf + XE;
  unsigned short* wkb = wqb + WE;
  unsigned short* wvb = wkb + WE;
  unsigned short* wob = wvb + WE;
  unsigned short* Qb  = wob + WE;      // (B,H,S,HD)
  unsigned short* Kb  = Qb  + XE;      // (B,H,S,HD)
  unsigned short* Vtb = Kb  + XE;      // (B,H,HD,S)
  unsigned short* ctx = Vtb + XE;      // (B,S,D)

  cvt_bf16_kernel<<<(int)((XE + 255) / 256), 256, 0, stream>>>(x,  xbf, (int)XE);
  cvt_bf16_kernel<<<(int)((WE + 255) / 256), 256, 0, stream>>>(Wq, wqb, (int)WE);
  cvt_bf16_kernel<<<(int)((WE + 255) / 256), 256, 0, stream>>>(Wk, wkb, (int)WE);
  cvt_bf16_kernel<<<(int)((WE + 255) / 256), 256, 0, stream>>>(Wv, wvb, (int)WE);
  cvt_bf16_kernel<<<(int)((WE + 255) / 256), 256, 0, stream>>>(Wo, wob, (int)WE);

  const int gemmBlocks = (MM / 32) * (DD / 32) / 8;  // 1024 blocks of 8 waves
  gemm_bf16_kernel<<<gemmBlocks, 256, 0, stream>>>(xbf, wqb, Qb,  0);
  gemm_bf16_kernel<<<gemmBlocks, 256, 0, stream>>>(xbf, wkb, Kb,  0);
  gemm_bf16_kernel<<<gemmBlocks, 256, 0, stream>>>(xbf, wvb, Vtb, 1);

  attn_kernel<<<dim3(SS / 64, BB * HH), 128, 0, stream>>>(Qb, Kb, Vtb, ctx);

  gemm_bf16_kernel<<<gemmBlocks, 256, 0, stream>>>(ctx, wob, d_out, 2);
}
